// LocalGlobalFusionCLIO_19207093748299
// MI455X (gfx1250) — matrix operand order repaired and checked by reading, not compiled, not run on hardware
//
#include <hip/hip_runtime.h>
#include <hip/hip_bf16.h>
#include <math.h>
#include <stdint.h>

// ---------------------------------------------------------------------------
// Model constants
// ---------------------------------------------------------------------------
#define B_    8
#define N_    8192
#define NP_   1024
#define NP4_  512

typedef _Float16 v16h __attribute__((ext_vector_type(16)));
typedef float    v8f  __attribute__((ext_vector_type(8)));

__device__ __forceinline__ v16h pack16(float4 a, float4 b, float4 c, float4 d)
{
    v16h r;
    r[0]  = (_Float16)a.x; r[1]  = (_Float16)a.y; r[2]  = (_Float16)a.z; r[3]  = (_Float16)a.w;
    r[4]  = (_Float16)b.x; r[5]  = (_Float16)b.y; r[6]  = (_Float16)b.z; r[7]  = (_Float16)b.w;
    r[8]  = (_Float16)c.x; r[9]  = (_Float16)c.y; r[10] = (_Float16)c.z; r[11] = (_Float16)c.w;
    r[12] = (_Float16)d.x; r[13] = (_Float16)d.y; r[14] = (_Float16)d.z; r[15] = (_Float16)d.w;
    return r;
}

// ---------------------------------------------------------------------------
// Generic strided/batched WMMA GEMM:  C[o,p] = sum_k A[o,k]*B[k,p]  (+bias,+relu)
// One wave computes a 16x32 output (two 16x16 WMMA tiles sharing the A frag).
// Interior tiles are staged global->LDS with CDNA5 async-LDS loads
// (global_load_async_to_lds_b32, ASYNCcnt); edges use branchless clamped loads.
// Fragments are then read from LDS as aligned b128 loads in contiguous k order
// (k = 16*half + s, same for A and B) and fed to v_wmma_f32_16x16x32_f16.
// ---------------------------------------------------------------------------
__global__ void gemm_wmma_kernel(const float* __restrict__ A,
                                 const float* __restrict__ Bm,
                                 float* __restrict__ C,
                                 const float* __restrict__ bias, int relu,
                                 int O, int K, int P,
                                 int laO, int laC, long aOut, long aIn,
                                 int lbC, int lbP, long bOut, long bIn,
                                 int lcO, int lcP, long cOut, long cIn,
                                 int innerCnt, int TO, int tiles)
{
    __shared__ float As[4][16][32];      // [wave][m][k]
    __shared__ float Bt[4][32][36];      // [wave][p][k]  (pad 36 keeps 16B align + banks spread)

    int w = threadIdx.x >> 5;
    int t = blockIdx.x * 4 + w;
    if (t >= tiles) return;              // wave-uniform exit, no barriers used
    int to = t % TO;
    int tp = t / TO;
    int z = blockIdx.y;
    int outer = z / innerCnt, inner = z % innerCnt;
    const float* Ab = A  + outer * aOut + inner * aIn;
    const float* Bb = Bm + outer * bOut + inner * bIn;
    float*       Cb = C  + outer * cOut + inner * cIn;

    int lane = threadIdx.x & 31;
    int m16  = lane & 15;
    int half = lane >> 4;
    int obase = to * 16, pbase = tp * 32;

    v8f acc0 = {}, acc1 = {};
    for (int kb = 0; kb < K; kb += 32) {
        bool fullK = (kb + 32 <= K);
        // ---------------- stage A tile (16 rows x 32 k) ----------------
        if (fullK && (obase + 16 <= O)) {
            // async global->LDS copy, lane supplies k index (coalesced for laC==1)
            for (int j = 0; j < 16; ++j) {
                unsigned long long g = (unsigned long long)(uintptr_t)
                    (Ab + (long)(obase + j) * laO + (long)(kb + lane) * laC);
                unsigned l = (unsigned)(uintptr_t)&As[w][j][lane];
                asm volatile("global_load_async_to_lds_b32 %0, %1, off"
                             :: "v"(l), "v"(g) : "memory");
            }
        } else {
            for (int j = 0; j < 16; ++j) {
                int row = obase + j, gk = kb + lane;
                int cr = row < O ? row : O - 1;
                int ck = gk  < K ? gk  : K - 1;
                float v = Ab[(long)cr * laO + (long)ck * laC];
                As[w][j][lane] = (row < O && gk < K) ? v : 0.f;
            }
        }
        // ---------------- stage B tile (32 k x 32 p), transposed ----------------
        if (fullK && (pbase + 32 <= P)) {
            for (int j = 0; j < 32; ++j) {
                unsigned long long g = (unsigned long long)(uintptr_t)
                    (Bb + (long)(kb + j) * lbC + (long)(pbase + lane) * lbP);
                unsigned l = (unsigned)(uintptr_t)&Bt[w][lane][j];
                asm volatile("global_load_async_to_lds_b32 %0, %1, off"
                             :: "v"(l), "v"(g) : "memory");
            }
        } else {
            for (int j = 0; j < 32; ++j) {
                int gk = kb + j, p = pbase + lane;
                int ck = gk < K ? gk : K - 1;
                int cp = p  < P ? p  : P - 1;
                float v = Bb[(long)ck * lbC + (long)cp * lbP];
                Bt[w][lane][j] = (gk < K && p < P) ? v : 0.f;
            }
        }
        asm volatile("s_wait_asynccnt 0x0" ::: "memory");

        // ---------------- fragments from LDS (WMMA lane layout) ----------------
        // A: lane(m16,half) needs k run [half*16, +16)
        const float4* ra0 = (const float4*)&As[w][m16][half * 16];
        const float4* ra1 = (const float4*)&As[w][m16][half * 16 + 8];
        v16h av = pack16(ra0[0], ra0[1], ra1[0], ra1[1]);
        // B: lane(col,half) needs k run [half*16, +16) of its column
        const float4* rb0 = (const float4*)&Bt[w][m16][half * 16];
        const float4* rb1 = (const float4*)&Bt[w][16 + m16][half * 16];
        v16h bv0 = pack16(rb0[0], rb0[1], rb0[2], rb0[3]);
        v16h bv1 = pack16(rb1[0], rb1[1], rb1[2], rb1[3]);

        acc0 = __builtin_amdgcn_wmma_f32_16x16x32_f16(false, av, false, bv0,
                                                      (short)0, acc0, false, false);
        acc1 = __builtin_amdgcn_wmma_f32_16x16x32_f16(false, av, false, bv1,
                                                      (short)0, acc1, false, false);
    }
    // ---- D layout: vgpr i -> row i (lanes 0-15) / row 8+i (lanes 16-31) ----
#pragma unroll
    for (int i = 0; i < 8; ++i) {
        int o = obase + i + half * 8;
        if (o < O) {
            int p0 = pbase + m16;
            int p1 = pbase + 16 + m16;
            if (p0 < P) {
                float v = acc0[i];
                if (bias) v += bias[o];
                if (relu) v = v > 0.f ? v : 0.f;
                Cb[(long)o * lcO + (long)p0 * lcP] = v;
            }
            if (p1 < P) {
                float v = acc1[i];
                if (bias) v += bias[o];
                if (relu) v = v > 0.f ? v : 0.f;
                Cb[(long)o * lcO + (long)p1 * lcP] = v;
            }
        }
    }
}

// ---------------------------------------------------------------------------
// Farthest point sampling: one block per batch, dist vector resident in LDS.
// ---------------------------------------------------------------------------
__global__ void fps_kernel(const float* __restrict__ xyz, int n, int npoint,
                           int* __restrict__ out)
{
    int b = blockIdx.x;
    const float* X = xyz + (long)b * n * 3;
    __shared__ float dist[8192];
    __shared__ float rv[256];
    __shared__ int   ri[256];
    __shared__ int   sFar;
    for (int i = threadIdx.x; i < n; i += blockDim.x) dist[i] = 1e10f;
    if (threadIdx.x == 0) sFar = 0;
    __syncthreads();
    for (int s = 0; s < npoint; ++s) {
        int far = sFar;
        if (threadIdx.x == 0) out[b * npoint + s] = far;
        float cx = X[far * 3], cy = X[far * 3 + 1], cz = X[far * 3 + 2];
        float bm = -1.f; int bi = 0;
        for (int i = threadIdx.x; i < n; i += blockDim.x) {
            float dx = X[i * 3] - cx, dy = X[i * 3 + 1] - cy, dz = X[i * 3 + 2] - cz;
            float d = dx * dx + dy * dy + dz * dz;
            float dd = dist[i];
            if (d < dd) dd = d;
            dist[i] = dd;
            if (dd > bm) { bm = dd; bi = i; }
        }
        rv[threadIdx.x] = bm; ri[threadIdx.x] = bi;
        __syncthreads();
        for (int stp = 128; stp > 0; stp >>= 1) {
            if ((int)threadIdx.x < stp) {
                float ov = rv[threadIdx.x + stp]; int oi = ri[threadIdx.x + stp];
                if (ov > rv[threadIdx.x] || (ov == rv[threadIdx.x] && oi < ri[threadIdx.x])) {
                    rv[threadIdx.x] = ov; ri[threadIdx.x] = oi;
                }
            }
            __syncthreads();
        }
        if (threadIdx.x == 0) sFar = ri[0];
        __syncthreads();
    }
}

__global__ void gather_xyz_kernel(const float* __restrict__ xyz,
                                  const int* __restrict__ fidx,
                                  float* __restrict__ nxyz, int n, int np)
{
    int t = blockIdx.x * blockDim.x + threadIdx.x;
    if (t >= B_ * np) return;
    int b = t / np, j = t % np;
    int i = fidx[b * np + j];
    const float* s = xyz + ((long)b * n + i) * 3;
    float* d = nxyz + ((long)b * np + j) * 3;
    d[0] = s[0]; d[1] = s[1]; d[2] = s[2];
}

// Ball query + grouping: g[b][c][j*K+k]; rows 0..2 relative xyz, 3.. feats
__global__ void ball_group_kernel(const float* __restrict__ xyz,
                                  const float* __restrict__ feats,
                                  const float* __restrict__ nxyz,
                                  float* __restrict__ g,
                                  int n, int np, int K, int Cin, float r2)
{
    int t = blockIdx.x * blockDim.x + threadIdx.x;
    if (t >= B_ * np) return;
    int b = t / np, j = t % np;
    const float* X  = xyz + (long)b * n * 3;
    const float* Fe = feats + (long)b * Cin * n;
    const float* c0 = nxyz + ((long)b * np + j) * 3;
    float cx = c0[0], cy = c0[1], cz = c0[2];
    int idxs[32];
    int cnt = 0;
    for (int i = 0; i < n && cnt < K; ++i) {
        float dx = X[i * 3] - cx, dy = X[i * 3 + 1] - cy, dz = X[i * 3 + 2] - cz;
        float d = dx * dx + dy * dy + dz * dz;
        if (d < r2) idxs[cnt++] = i;
    }
    int first = (cnt > 0) ? idxs[0] : 0;
    for (int k = cnt; k < K; ++k) idxs[k] = first;
    long P = (long)np * K;
    float* G = g + (long)b * (3 + Cin) * P;
    for (int k = 0; k < K; ++k) {
        int i = idxs[k];
        long o = (long)j * K + k;
        G[0 * P + o] = X[i * 3]     - cx;
        G[1 * P + o] = X[i * 3 + 1] - cy;
        G[2 * P + o] = X[i * 3 + 2] - cz;
        for (int c = 0; c < Cin; ++c) G[(3 + c) * P + o] = Fe[(long)c * n + i];
    }
}

// BatchNorm stats per channel over (batch, spatial)
__global__ void bn_stats_kernel(const float* __restrict__ X, float* __restrict__ mean,
                                float* __restrict__ rstd, long P, long batchStride, int nb)
{
    int c = blockIdx.x;
    __shared__ float s1[256], s2[256];
    float a = 0.f, b = 0.f;
    for (int bb = 0; bb < nb; ++bb) {
        const float* p = X + (long)bb * batchStride + (long)c * P;
        for (long i = threadIdx.x; i < P; i += 256) { float v = p[i]; a += v; b += v * v; }
    }
    s1[threadIdx.x] = a; s2[threadIdx.x] = b;
    __syncthreads();
    for (int s = 128; s > 0; s >>= 1) {
        if ((int)threadIdx.x < s) { s1[threadIdx.x] += s1[threadIdx.x + s]; s2[threadIdx.x] += s2[threadIdx.x + s]; }
        __syncthreads();
    }
    if (threadIdx.x == 0) {
        float cnt = (float)((long)nb * P);
        float m = s1[0] / cnt;
        float v = s2[0] / cnt - m * m;
        mean[c] = m;
        rstd[c] = rsqrtf(fmaxf(v, 0.f) + 1e-5f);
    }
}

__global__ void bn_apply_kernel(float* __restrict__ X, const float* __restrict__ mean,
                                const float* __restrict__ rstd, const float* __restrict__ gam,
                                const float* __restrict__ bet, int C, long P, long total)
{
    long i = (long)blockIdx.x * blockDim.x + threadIdx.x;
    if (i >= total) return;
    long r = i % ((long)C * P);
    int c = (int)(r / P);
    float v = (X[i] - mean[c]) * rstd[c] * gam[c] + bet[c];
    X[i] = v > 0.f ? v : 0.f;
}

__global__ void maxpool_kernel(const float* __restrict__ X, float* __restrict__ out,
                               int C, int np, int K, int cOff, int ldC)
{
    long t = (long)blockIdx.x * blockDim.x + threadIdx.x;
    long total = (long)B_ * C * np;
    if (t >= total) return;
    int b = (int)(t / ((long)C * np));
    long r = t % ((long)C * np);
    int c = (int)(r / np), j = (int)(r % np);
    const float* p = X + (((long)b * C + c) * np + j) * K;
    float m = p[0];
    for (int k = 1; k < K; ++k) m = fmaxf(m, p[k]);
    out[((long)b * ldC + cOff + c) * np + j] = m;
}

__global__ void knn3_kernel(const float* __restrict__ ux, const float* __restrict__ kx,
                            int* __restrict__ idx3, float* __restrict__ w3, int nu, int nk)
{
    int t = blockIdx.x * blockDim.x + threadIdx.x;
    if (t >= B_ * nu) return;
    int b = t / nu, i = t % nu;
    const float* U = ux + ((long)b * nu + i) * 3;
    const float* Kx = kx + (long)b * nk * 3;
    float ux0 = U[0], ux1 = U[1], ux2 = U[2];
    float d0 = 1e30f, d1 = 1e30f, d2 = 1e30f;
    int i0 = 0, i1 = 0, i2 = 0;
    for (int j = 0; j < nk; ++j) {
        float dx = Kx[j * 3] - ux0, dy = Kx[j * 3 + 1] - ux1, dz = Kx[j * 3 + 2] - ux2;
        float d = dx * dx + dy * dy + dz * dz;
        if (d < d0)      { d2 = d1; i2 = i1; d1 = d0; i1 = i0; d0 = d; i0 = j; }
        else if (d < d1) { d2 = d1; i2 = i1; d1 = d; i1 = j; }
        else if (d < d2) { d2 = d; i2 = j; }
    }
    float w0 = 1.f / (d0 + 1e-8f), w1 = 1.f / (d1 + 1e-8f), w2 = 1.f / (d2 + 1e-8f);
    float s = w0 + w1 + w2;
    long o = ((long)b * nu + i) * 3;
    idx3[o] = i0; idx3[o + 1] = i1; idx3[o + 2] = i2;
    w3[o] = w0 / s; w3[o + 1] = w1 / s; w3[o + 2] = w2 / s;
}

__global__ void interp_kernel(const float* __restrict__ kf, const int* __restrict__ idx3,
                              const float* __restrict__ w3, float* __restrict__ out,
                              int Ck, int nk, int nu, int Ctot)
{
    long t = (long)blockIdx.x * blockDim.x + threadIdx.x;
    long total = (long)B_ * Ck * nu;
    if (t >= total) return;
    int b = (int)(t / ((long)Ck * nu));
    long r = t % ((long)Ck * nu);
    int c = (int)(r / nu), i = (int)(r % nu);
    const float* F = kf + ((long)b * Ck + c) * nk;
    long o = ((long)b * nu + i) * 3;
    out[((long)b * Ctot + c) * nu + i] =
        w3[o] * F[idx3[o]] + w3[o + 1] * F[idx3[o + 1]] + w3[o + 2] * F[idx3[o + 2]];
}

__global__ void copy_rows_kernel(const float* __restrict__ src, float* __restrict__ dst,
                                 int Cs, int n, int rowOff, int Cd)
{
    long t = (long)blockIdx.x * blockDim.x + threadIdx.x;
    long total = (long)B_ * Cs * n;
    if (t >= total) return;
    int b = (int)(t / ((long)Cs * n));
    long r = t % ((long)Cs * n);
    int c = (int)(r / n), i = (int)(r % n);
    dst[((long)b * Cd + rowOff + c) * n + i] = src[((long)b * Cs + c) * n + i];
}

__global__ void softmax_kernel(float* __restrict__ a, int n, float scale, int rows)
{
    int t = blockIdx.x * blockDim.x + threadIdx.x;
    if (t >= rows) return;
    float* p = a + (long)t * n;
    float m = -1e30f;
    for (int j = 0; j < n; ++j) { float v = p[j] * scale; if (v > m) m = v; }
    float s = 0.f;
    for (int j = 0; j < n; ++j) { float e = __expf(p[j] * scale - m); p[j] = e; s += e; }
    float inv = 1.f / s;
    for (int j = 0; j < n; ++j) p[j] *= inv;
}

// residual add + LayerNorm over channel axis
__global__ void resid_ln_kernel(const float* __restrict__ y, const float* __restrict__ x,
                                const float* __restrict__ g, const float* __restrict__ be,
                                float* __restrict__ out, int C, int n)
{
    int t = blockIdx.x * blockDim.x + threadIdx.x;
    if (t >= B_ * n) return;
    int b = t / n, i = t % n;
    long base = (long)b * C * n + i;
    float s = 0.f, ss = 0.f;
    for (int c = 0; c < C; ++c) {
        float v = y[base + (long)c * n] + x[base + (long)c * n];
        s += v; ss += v * v;
    }
    float m = s / C;
    float var = ss / C - m * m;
    float r = rsqrtf(fmaxf(var, 0.f) + 1e-5f);
    for (int c = 0; c < C; ++c) {
        float v = y[base + (long)c * n] + x[base + (long)c * n];
        out[base + (long)c * n] = (v - m) * r * g[c] + be[c];
    }
}

__global__ void chan_stats_kernel(const float* __restrict__ X, float* __restrict__ cs,
                                  int C, int n)
{
    int t = blockIdx.x * blockDim.x + threadIdx.x;
    if (t >= B_ * C) return;
    int b = t / C, c = t % C;
    const float* p = X + ((long)b * C + c) * n;
    float s = 0.f, ss = 0.f, mx = -1e30f, mn = 1e30f;
    for (int i = 0; i < n; ++i) {
        float v = p[i];
        s += v; ss += v * v;
        mx = fmaxf(mx, v); mn = fminf(mn, v);
    }
    float m = s / n;
    float var = (ss - n * m * m) / (n - 1);       // ddof=1
    long o = ((long)b * C + c) * 4;
    cs[o] = m; cs[o + 1] = sqrtf(fmaxf(var, 0.f)); cs[o + 2] = mx; cs[o + 3] = mn;
}

__global__ void reduce_stats_kernel(const float* __restrict__ cs, float* __restrict__ sb, int C)
{
    int b = blockIdx.x * blockDim.x + threadIdx.x;
    if (b >= B_) return;
    float s0 = 0, s1 = 0, s2 = 0, s3 = 0, s4 = 0;
    for (int c = 0; c < C; ++c) {
        long o = ((long)b * C + c) * 4;
        s0 += cs[o]; s1 += cs[o + 1]; s2 += cs[o + 2]; s3 += cs[o + 3];
        s4 += cs[o + 2] - cs[o + 3];
    }
    float inv = 1.f / C;
    sb[b * 5]     = s0 * inv; sb[b * 5 + 1] = s1 * inv; sb[b * 5 + 2] = s2 * inv;
    sb[b * 5 + 3] = s3 * inv; sb[b * 5 + 4] = s4 * inv;
}

__global__ void broadcast_stats_kernel(const float* __restrict__ sb, float* __restrict__ dst,
                                       int Ctot, int n, int rowOff)
{
    int t = blockIdx.x * blockDim.x + threadIdx.x;
    if (t >= B_ * 5 * n) return;
    int b = t / (5 * n);
    int r = t % (5 * n);
    int s = r / n, i = r % n;
    dst[((long)b * Ctot + rowOff + s) * n + i] = sb[b * 5 + s];
}

// ---------------------------------------------------------------------------
// Host-side helpers
// ---------------------------------------------------------------------------
struct MlpL { const float* W; const float* g; const float* b; int Cout; };

static void launch_gemm(hipStream_t st, const float* A, const float* Bm, float* C,
                        const float* bias, int relu, int O, int K, int P,
                        int laO, int laC, long aOut, long aIn,
                        int lbC, int lbP, long bOut, long bIn,
                        int lcO, int lcP, long cOut, long cIn,
                        int batches, int innerCnt)
{
    int TO = (O + 15) / 16, TP = (P + 31) / 32;   // 16x32 output per wave
    int tiles = TO * TP;
    dim3 grid((tiles + 3) / 4, batches);
    gemm_wmma_kernel<<<grid, 128, 0, st>>>(A, Bm, C, bias, relu, O, K, P,
                                           laO, laC, aOut, aIn,
                                           lbC, lbP, bOut, bIn,
                                           lcO, lcP, cOut, cIn,
                                           innerCnt, TO, tiles);
}

// Weight GEMM: Y[b] = W(OxK) * X[b](KxP)   (row-major X/Y with row stride P)
static void gemm_w(hipStream_t st, const float* W, const float* X, float* Y,
                   const float* bias, int relu, int O, int K, long P,
                   long bStride, long cStride)
{
    launch_gemm(st, W, X, Y, bias, relu, O, K, (int)P,
                K, 1, 0, 0,
                (int)P, 1, bStride, 0,
                (int)P, 1, cStride, 0,
                B_, 1);
}

static void bn_forward(hipStream_t st, float* Y, const float* gam, const float* bet,
                       int C, long P, float* stat)
{
    bn_stats_kernel<<<C, 256, 0, st>>>(Y, stat, stat + C, P, (long)C * P, B_);
    long tot = (long)B_ * C * P;
    bn_apply_kernel<<<(int)((tot + 255) / 256), 256, 0, st>>>(Y, stat, stat + C, gam, bet, C, P, tot);
}

static void run_sa(hipStream_t st, const float* xyz, int n, const float* feats, int Cin,
                   int np, const float* rr, const int* Ks,
                   const MlpL* s0, int n0, const MlpL* s1, int n1,
                   int* fidx, float* nxyz, float* cat, int Ctot,
                   float* bufA, float* bufB, float* stat)
{
    fps_kernel<<<B_, 256, 0, st>>>(xyz, n, np, fidx);
    int tg = B_ * np;
    gather_xyz_kernel<<<(tg + 255) / 256, 256, 0, st>>>(xyz, fidx, nxyz, n, np);
    const MlpL* sc[2] = { s0, s1 };
    int nl[2] = { n0, n1 };
    int coff = 0;
    for (int s = 0; s < 2; ++s) {
        int K = Ks[s];
        long P = (long)np * K;
        ball_group_kernel<<<(tg + 255) / 256, 256, 0, st>>>(xyz, feats, nxyz, bufA,
                                                            n, np, K, Cin, rr[s] * rr[s]);
        float* cur = bufA; float* nxt = bufB; int curC = 3 + Cin;
        for (int l = 0; l < nl[s]; ++l) {
            const MlpL& L = sc[s][l];
            gemm_w(st, L.W, cur, nxt, nullptr, 0, L.Cout, curC, P,
                   (long)curC * P, (long)L.Cout * P);
            bn_forward(st, nxt, L.g, L.b, L.Cout, P, stat);
            float* tswap = cur; cur = nxt; nxt = tswap; curC = L.Cout;
        }
        long tm = (long)B_ * curC * np;
        maxpool_kernel<<<(int)((tm + 255) / 256), 256, 0, st>>>(cur, cat, curC, np, K, coff, Ctot);
        coff += curC;
    }
}

static void run_fp(hipStream_t st, const float* ux, int nu, const float* kx, int nk,
                   const float* kf, int Ck, const float* uf, int Cu,
                   const float* W, const float* gam, const float* bet, int Cout,
                   float* out, int* idx3, float* w3, float* bufA, float* stat)
{
    int tot = B_ * nu;
    knn3_kernel<<<(tot + 255) / 256, 256, 0, st>>>(ux, kx, idx3, w3, nu, nk);
    int Ctot = Ck + Cu;
    long t2 = (long)B_ * Ck * nu;
    interp_kernel<<<(int)((t2 + 255) / 256), 256, 0, st>>>(kf, idx3, w3, bufA, Ck, nk, nu, Ctot);
    if (uf) {
        long t3 = (long)B_ * Cu * nu;
        copy_rows_kernel<<<(int)((t3 + 255) / 256), 256, 0, st>>>(uf, bufA, Cu, nu, Ck, Ctot);
    }
    gemm_w(st, W, bufA, out, nullptr, 0, Cout, Ctot, nu, (long)Ctot * nu, (long)Cout * nu);
    bn_forward(st, out, gam, bet, Cout, nu, stat);
}

// ---------------------------------------------------------------------------
extern "C" void kernel_launch(void* const* d_in, const int* in_sizes, int n_in,
                              void* d_out, int out_size, void* d_ws, size_t ws_size,
                              hipStream_t stream)
{
    (void)in_sizes; (void)n_in; (void)out_size; (void)ws_size;
    auto F = [&](int i) { return (const float*)d_in[i]; };

    const float* pc  = F(0);                     // (8,8192,3)
    const float* l0f = F(1);                     // (8,8192,1) == (8,1,8192) flat

    // pytree-flattened params (dict keys sorted)
    const float* attn_ln_b  = F(2);  const float* attn_ln_g  = F(3);
    const float* attn_pj_b  = F(4);  const float* attn_pj_w  = F(5);
    const float* attn_qkv_b = F(6);  const float* attn_qkv_w = F(7);
    const float* fp1W = F(8),  *fp1g = F(9),  *fp1b = F(10);
    const float* fp2W = F(11), *fp2g = F(12), *fp2b = F(13);
    const float* fp3W = F(14), *fp3g = F(15), *fp3b = F(16);
    const float* fp4W = F(17), *fp4g = F(18), *fp4b = F(19);
    const float* fusW = F(20), *fusB = F(21);
    const float* lin1W = F(22), *lin1b = F(23);
    const float* lin2W = F(24), *lin2b = F(25);
    const float* lin3W = F(26), *lin3b = F(27);
    const float* lin4W = F(28), *lin4b = F(29);
    MlpL sa1a[3] = { {F(30),F(31),F(32),32}, {F(33),F(34),F(35),32}, {F(36),F(37),F(38),64} };
    MlpL sa1b[3] = { {F(39),F(40),F(41),32}, {F(42),F(43),F(44),32}, {F(45),F(46),F(47),64} };
    MlpL sa2a[2] = { {F(48),F(49),F(50),64},  {F(51),F(52),F(53),64}  };
    MlpL sa2b[2] = { {F(54),F(55),F(56),64},  {F(57),F(58),F(59),128} };
    MlpL sa3a[2] = { {F(60),F(61),F(62),128}, {F(63),F(64),F(65),128} };
    MlpL sa3b[2] = { {F(66),F(67),F(68),128}, {F(69),F(70),F(71),128} };
    MlpL sa4a[2] = { {F(72),F(73),F(74),128}, {F(75),F(76),F(77),128} };
    MlpL sa4b[2] = { {F(78),F(79),F(80),128}, {F(81),F(82),F(83),128} };

    // ---- carve workspace -------------------------------------------------
    size_t off = 0;
    auto carve = [&](size_t elems, size_t elsz) -> void* {
        off = (off + 255) & ~(size_t)255;
        void* p = (char*)d_ws + off;
        off += elems * elsz;
        return p;
    };
    int*   fidx  = (int*)  carve((size_t)B_ * 1024, 4);
    float* l1xyz = (float*)carve((size_t)B_ * 1024 * 3, 4);
    float* l2xyz = (float*)carve((size_t)B_ * 1024 * 3, 4);
    float* l3xyz = (float*)carve((size_t)B_ * 1024 * 3, 4);
    float* l4xyz = (float*)carve((size_t)B_ * 512  * 3, 4);
    float* cat   = (float*)carve((size_t)B_ * 256 * 1024, 4);
    float* l1    = (float*)carve((size_t)B_ * 64  * 1024, 4);
    float* l2    = (float*)carve((size_t)B_ * 128 * 1024, 4);
    float* l3    = (float*)carve((size_t)B_ * 128 * 1024, 4);
    float* l4    = (float*)carve((size_t)B_ * 256 * 512, 4);
    float* stat  = (float*)carve(512, 4);
    float* qkvB  = (float*)carve((size_t)B_ * 768 * 512, 4);
    float* scores= (float*)carve((size_t)8 * 512 * 512, 4);     // per-batch, 8 heads
    float* obuf  = (float*)carve((size_t)B_ * 256 * 512, 4);
    float* attnO = (float*)carve((size_t)B_ * 256 * 512, 4);
    float* l4n   = (float*)carve((size_t)B_ * 256 * 512, 4);
    float* cs    = (float*)carve((size_t)B_ * 256 * 4, 4);
    float* sb    = (float*)carve((size_t)B_ * 5, 4);
    float* fusIn = (float*)carve((size_t)B_ * 261 * 512, 4);
    float* fused = (float*)carve((size_t)B_ * 256 * 512, 4);
    float* l3n   = (float*)carve((size_t)B_ * 256 * 1024, 4);
    float* l2n   = (float*)carve((size_t)B_ * 128 * 1024, 4);
    float* l1n   = (float*)carve((size_t)B_ * 128 * 1024, 4);
    int*   idx3  = (int*)  carve((size_t)B_ * 8192 * 3, 4);
    float* w3    = (float*)carve((size_t)B_ * 8192 * 3, 4);
    float* bufA  = (float*)carve((size_t)B_ * 131 * 32768, 4);  // grouped / concat arena
    float* bufB  = (float*)carve((size_t)B_ * 131 * 32768, 4);  // MLP ping-pong arena

    // ---- SA layers -------------------------------------------------------
    { float r[2] = {2.f, 4.f};  int K[2] = {4, 8};
      run_sa(stream, pc, N_, l0f, 1, NP_, r, K, sa1a, 3, sa1b, 3,
             fidx, l1xyz, cat, 128, bufA, bufB, stat); }
    gemm_w(stream, lin1W, cat, l1, lin1b, 0, 64, 128, 1024, 128L * 1024, 64L * 1024);

    { float r[2] = {4.f, 8.f};  int K[2] = {8, 16};
      run_sa(stream, l1xyz, 1024, l1, 64, NP_, r, K, sa2a, 2, sa2b, 2,
             fidx, l2xyz, cat, 192, bufA, bufB, stat); }
    gemm_w(stream, lin2W, cat, l2, lin2b, 0, 128, 192, 1024, 192L * 1024, 128L * 1024);

    { float r[2] = {8.f, 16.f}; int K[2] = {16, 32};
      run_sa(stream, l2xyz, 1024, l2, 128, NP_, r, K, sa3a, 2, sa3b, 2,
             fidx, l3xyz, cat, 256, bufA, bufB, stat); }
    gemm_w(stream, lin3W, cat, l3, lin3b, 0, 128, 256, 1024, 256L * 1024, 128L * 1024);

    { float r[2] = {16.f, 32.f}; int K[2] = {16, 32};
      run_sa(stream, l3xyz, 1024, l3, 128, NP4_, r, K, sa4a, 2, sa4b, 2,
             fidx, l4xyz, cat, 256, bufA, bufB, stat); }
    gemm_w(stream, lin4W, cat, l4, lin4b, 0, 256, 256, 512, 256L * 512, 256L * 512);

    // ---- attention (H=8, hd=32, n=512) -----------------------------------
    gemm_w(stream, attn_qkv_w, l4, qkvB, attn_qkv_b, 0, 768, 256, 512,
           256L * 512, 768L * 512);
    const float scl = 0.17677669529663687f;    // 1/sqrt(32)
    for (int b = 0; b < B_; ++b) {
        const float* qb = qkvB + (long)b * 768 * 512;
        const float* kb = qb + 256 * 512;
        const float* vb = qb + 2 * 256 * 512;
        // scores[h][i][j] = sum_d q[h][d][i] * k[h][d][j]
        launch_gemm(stream, qb, kb, scores, nullptr, 0, 512, 32, 512,
                    1, 512, 0, 32L * 512,
                    512, 1, 0, 32L * 512,
                    512, 1, 0, 512L * 512,
                    8, 8);
        int rows = 8 * 512;
        softmax_kernel<<<(rows + 255) / 256, 256, 0, stream>>>(scores, 512, scl, rows);
        // o[h][d][i] = sum_j v[h][d][j] * a[h][i][j]
        launch_gemm(stream, vb, scores, obuf + (long)b * 256 * 512, nullptr, 0, 32, 512, 512,
                    512, 1, 0, 32L * 512,
                    1, 512, 0, 512L * 512,
                    512, 1, 0, 32L * 512,
                    8, 8);
    }
    gemm_w(stream, attn_pj_w, obuf, attnO, attn_pj_b, 0, 256, 256, 512,
           256L * 512, 256L * 512);
    {
        int tl = B_ * 512;
        resid_ln_kernel<<<(tl + 255) / 256, 256, 0, stream>>>(attnO, l4, attn_ln_g, attn_ln_b,
                                                              l4n, 256, 512);
    }

    // ---- global stats + fusion -------------------------------------------
    {
        int tc = B_ * 256;
        chan_stats_kernel<<<(tc + 255) / 256, 256, 0, stream>>>(l4n, cs, 256, 512);
        reduce_stats_kernel<<<1, B_, 0, stream>>>(cs, sb, 256);
        long t3 = (long)B_ * 256 * 512;
        copy_rows_kernel<<<(int)((t3 + 255) / 256), 256, 0, stream>>>(l4n, fusIn, 256, 512, 0, 261);
        int tb = B_ * 5 * 512;
        broadcast_stats_kernel<<<(tb + 255) / 256, 256, 0, stream>>>(sb, fusIn, 261, 512, 256);
        gemm_w(stream, fusW, fusIn, fused, fusB, 0, 256, 261, 512, 261L * 512, 256L * 512);
    }

    // ---- feature propagation ---------------------------------------------
    run_fp(stream, l3xyz, 1024, l4xyz, 512, fused, 256, l3, 128,
           fp4W, fp4g, fp4b, 256, l3n, idx3, w3, bufA, stat);
    run_fp(stream, l2xyz, 1024, l3xyz, 1024, l3n, 256, l2, 128,
           fp3W, fp3g, fp3b, 128, l2n, idx3, w3, bufA, stat);
    run_fp(stream, l1xyz, 1024, l2xyz, 1024, l2n, 128, l1, 64,
           fp2W, fp2g, fp2b, 128, l1n, idx3, w3, bufA, stat);
    // fp1 writes straight into d_out after the l4_xyz block
    float* l0out = (float*)d_out + (size_t)B_ * 512 * 3;
    run_fp(stream, pc, 8192, l1xyz, 1024, l1n, 128, nullptr, 0,
           fp1W, fp1g, fp1b, 128, l0out, idx3, w3, bufA, stat);

    // ---- emit l4_xyz ------------------------------------------------------
    hipMemcpyAsync(d_out, l4xyz, (size_t)B_ * 512 * 3 * sizeof(float),
                   hipMemcpyDeviceToDevice, stream);
}
